// RNN_87299505259276
// MI455X (gfx1250) — matrix operand order repaired and checked by reading, not compiled, not run on hardware
//
#include <hip/hip_runtime.h>

// ---------------- problem constants ----------------
#define SEQ   256
#define BATCH 64
#define XD    64
#define FEAT  256
#define DEC   256
#define HDIM  512
#define G4    2048   // 4*HDIM

// ---------------- CDNA5 WMMA types ----------------
typedef __attribute__((ext_vector_type(16))) __bf16 v16bf;
typedef __attribute__((ext_vector_type(8)))  float  v8f;

union ABFrag {
  unsigned short s[16];
  uint4          q[2];
  v16bf          v;
};

__device__ __forceinline__ unsigned short f32_to_bf16(float f) {
  unsigned int u = __float_as_uint(f);
  u += 0x7FFFu + ((u >> 16) & 1u);   // round-to-nearest-even
  return (unsigned short)(u >> 16);
}
__device__ __forceinline__ float bf16_to_f32(unsigned short h) {
  return __uint_as_float(((unsigned int)h) << 16);
}

// Branch-free activations: v_exp_f32 (TRANS, co-executes with WMMA) + v_rcp_f32.
__device__ __forceinline__ float fast_tanh(float x) {
  float xc = fminf(fmaxf(x, -10.0f), 10.0f);   // tanh(±10) == ±1 in f32
  float e = __expf(2.0f * xc);
  return (e - 1.0f) * __builtin_amdgcn_rcpf(e + 1.0f);
}
__device__ __forceinline__ float fast_sigmoid(float x) {
  return __builtin_amdgcn_rcpf(1.0f + __expf(-x));
}

__device__ __forceinline__ v8f wmma_bf16(v16bf a, v16bf b, v8f c) {
  // (neg_a, A, neg_b, B, c_mod, C, reuse_a, reuse_b)
  return __builtin_amdgcn_wmma_f32_16x16x32_bf16(false, a, false, b, (short)0, c,
                                                 false, false);
}

// A fragment (16x32 bf16) from LDS, row-major activations with given stride.
// Lane L: row m0+(L&15); halves j<8 -> K=kb+8g+j ; j>=8 -> K=kb+16+8g+(j-8).
__device__ __forceinline__ v16bf load_a_lds(const unsigned short* base, int stride,
                                            int m0, int kb, int lane) {
  int m = m0 + (lane & 15);
  int g = lane >> 4;
  const unsigned short* p = base + m * stride + kb + 8 * g;
  ABFrag fr;
  fr.q[0] = *reinterpret_cast<const uint4*>(p);
  fr.q[1] = *reinterpret_cast<const uint4*>(p + 16);
  return fr.v;
}

// B fragment (32x16 bf16) from global bf16 weight matrix W[N][K] (row-major),
// implementing B[k][n] = W[n][k]. Lane L: col n0+(L&15); j<8 -> K=kb+8g+j ; j>=8 -> K=kb+16+8g+(j-8).
__device__ __forceinline__ v16bf load_b_w(const unsigned short* W, int stride,
                                          int n0, int kb, int lane) {
  int n = n0 + (lane & 15);
  int g = lane >> 4;
  const unsigned short* p = W + n * stride + kb + 8 * g;
  ABFrag fr;
  fr.q[0] = *reinterpret_cast<const uint4*>(p);
  fr.q[1] = *reinterpret_cast<const uint4*>(p + 16);
  return fr.v;
}

// A fragment from global f32 (converted to bf16 on the fly).
__device__ __forceinline__ v16bf load_a_g32(const float* base, int stride,
                                            int m0, int kb, int lane) {
  int m = m0 + (lane & 15);
  int g = lane >> 4;
  const float* p = base + m * stride + kb + 8 * g;
  ABFrag fr;
#pragma unroll
  for (int j = 0; j < 8; ++j) fr.s[j] = f32_to_bf16(p[j]);
#pragma unroll
  for (int j = 0; j < 8; ++j) fr.s[8 + j] = f32_to_bf16(p[16 + j]);
  return fr.v;
}

// ---------------- kernel 1: f32 -> bf16 weight conversion ----------------
__global__ void cvt_bf16_kernel(const float* __restrict__ src,
                                unsigned short* __restrict__ dst, int n) {
  int i = blockIdx.x * blockDim.x + threadIdx.x;
  if (i < n) dst[i] = f32_to_bf16(src[i]);
}

// ---------------- kernel 2: teacher-forced features ----------------
// feat_tf[S*B][FEAT] = tanh(x[S*B][XD] @ Wfx^T + bfx), one 16x16 tile per wave.
__global__ void feat_kernel(const float* __restrict__ x,
                            const unsigned short* __restrict__ Wfx_bf,
                            const float* __restrict__ bfx,
                            unsigned short* __restrict__ feat_tf) {
  int lane = threadIdx.x & 31;
  int gw = blockIdx.x * (blockDim.x >> 5) + (threadIdx.x >> 5);
  int rt = gw >> 4;   // 1024 row tiles over 16384 rows
  int ct = gw & 15;   // 16 col tiles over FEAT=256
  int m0 = rt * 16, n0 = ct * 16;
  int ln = lane & 15, gg = lane >> 4;
  int n = n0 + ln;

  v8f acc;
  float bn = bfx[n];
#pragma unroll
  for (int r = 0; r < 8; ++r) acc[r] = bn;

#pragma unroll
  for (int kb = 0; kb < XD; kb += 32)
    acc = wmma_bf16(load_a_g32(x, XD, m0, kb, lane),
                    load_b_w(Wfx_bf, XD, n0, kb, lane), acc);

#pragma unroll
  for (int r = 0; r < 8; ++r) {
    int m = m0 + r + 8 * gg;
    feat_tf[m * FEAT + n] = f32_to_bf16(fast_tanh(acc[r]));
  }
}

// ---------------- kernel 3: persistent sequential LSTM scan ----------------
// Single workgroup (32 wave32 waves on one WGP); all recurrent state in
// LDS + VGPRs; weights stream from L2 as bf16; all GEMMs via v_wmma bf16
// with dual accumulation chains sharing each A fragment (hides the
// WMMA->WMMA hazard slots and halves LDS A traffic).
__global__ __launch_bounds__(1024, 1)
void scan_kernel(const unsigned short* __restrict__ Wfx_bf,
                 const unsigned short* __restrict__ Wih_bf,
                 const unsigned short* __restrict__ Whh_bf,
                 const unsigned short* __restrict__ Whx_bf,
                 const unsigned short* __restrict__ Wout_bf,
                 const unsigned short* __restrict__ feat_tf,
                 const float* __restrict__ bfx,
                 const float* __restrict__ bih,
                 const float* __restrict__ bhh,
                 const float* __restrict__ bhx,
                 const float* __restrict__ bout,
                 const float* __restrict__ mode_sel,
                 float* __restrict__ out) {
  __shared__ __align__(16) unsigned short h_bf[BATCH * HDIM];    // 64 KB
  __shared__ __align__(16) unsigned short feat_bf[BATCH * FEAT]; // 32 KB
  __shared__ __align__(16) unsigned short dec_bf[BATCH * DEC];   // 32 KB
  __shared__ __align__(16) unsigned short y_bf[BATCH * XD];      //  8 KB
  __shared__ float bias_g[G4];                                   //  8 KB
  __shared__ float bias_hx[DEC];
  __shared__ float bias_fx[FEAT];
  __shared__ float bias_out[XD];

  const int tid = threadIdx.x;
  const int lane = tid & 31;
  const int wid = tid >> 5;      // 0..31
  const int ln = lane & 15;
  const int gg = lane >> 4;

  for (int i = tid; i < BATCH * HDIM; i += 1024) h_bf[i] = 0;
  for (int i = tid; i < BATCH * XD; i += 1024) y_bf[i] = 0;
  for (int i = tid; i < G4; i += 1024) bias_g[i] = bih[i] + bhh[i];
  if (tid < DEC) bias_hx[tid] = bhx[tid];
  if (tid < FEAT) bias_fx[tid] = bfx[tid];
  if (tid < XD) bias_out[tid] = bout[tid];
  __syncthreads();

  // cell state c: 4 (m,n) tiles per wave, f32 C/D fragment layout in VGPRs
  v8f c_acc[4];
#pragma unroll
  for (int p = 0; p < 4; ++p)
#pragma unroll
    for (int r = 0; r < 8; ++r) c_acc[p][r] = 0.0f;

  // this wave's two 16x16 tiles over [64 x 256] share the same row block
  const int m0_2 = ((wid * 2) >> 4) * 16;       // row tile (same for q=0,1)
  const int n0_2 = ((wid * 2) & 15) * 16;       // first col tile; second is +16

  for (int t = 0; t < SEQ; ++t) {
    const float mix = mode_sel[t];

    // ---- Phase 1: feat = blend(tanh(y_prev@Wfx^T+bfx), feat_tf[t]) ----
    {
      v8f acc0, acc1;
      float b0 = bias_fx[n0_2 + ln], b1 = bias_fx[n0_2 + 16 + ln];
#pragma unroll
      for (int r = 0; r < 8; ++r) { acc0[r] = b0; acc1[r] = b1; }
#pragma unroll
      for (int kb = 0; kb < XD; kb += 32) {
        v16bf a = load_a_lds(y_bf, XD, m0_2, kb, lane);
        acc0 = wmma_bf16(a, load_b_w(Wfx_bf, XD, n0_2, kb, lane), acc0);
        acc1 = wmma_bf16(a, load_b_w(Wfx_bf, XD, n0_2 + 16, kb, lane), acc1);
      }
#pragma unroll
      for (int q = 0; q < 2; ++q) {
        v8f acc = q ? acc1 : acc0;
        int n = n0_2 + q * 16 + ln;
#pragma unroll
        for (int r = 0; r < 8; ++r) {
          int m = m0_2 + r + 8 * gg;
          float fa = fast_tanh(acc[r]);
          float ftf = bf16_to_f32(feat_tf[(t * BATCH + m) * FEAT + n]);
          float fv = (t == 0) ? ftf : (mix * fa + (1.0f - mix) * ftf);
          feat_bf[m * FEAT + n] = f32_to_bf16(fv);
        }
      }
    }
    __syncthreads();

    // ---- Phase 2a: dec = tanh(h @ Whx^T + bhx) ----
    {
      v8f acc0, acc1;
      float b0 = bias_hx[n0_2 + ln], b1 = bias_hx[n0_2 + 16 + ln];
#pragma unroll
      for (int r = 0; r < 8; ++r) { acc0[r] = b0; acc1[r] = b1; }
      for (int kb = 0; kb < HDIM; kb += 32) {
        v16bf a = load_a_lds(h_bf, HDIM, m0_2, kb, lane);
        acc0 = wmma_bf16(a, load_b_w(Whx_bf, HDIM, n0_2, kb, lane), acc0);
        acc1 = wmma_bf16(a, load_b_w(Whx_bf, HDIM, n0_2 + 16, kb, lane), acc1);
      }
#pragma unroll
      for (int q = 0; q < 2; ++q) {
        v8f acc = q ? acc1 : acc0;
        int n = n0_2 + q * 16 + ln;
#pragma unroll
        for (int r = 0; r < 8; ++r)
          dec_bf[(m0_2 + r + 8 * gg) * DEC + n] = f32_to_bf16(fast_tanh(acc[r]));
      }
    }

    // ---- Phase 2b: gates = [feat|h] @ [Wih|Whh]^T + b ; update c ----
    // Gate pairs share one A fragment per k-step: (i,g) then (f,o).
    v8f h2v[4];
#pragma unroll
    for (int p = 0; p < 4; ++p) {
      int idx = wid * 4 + p;               // 128 (m,n) pairs: 4 x 32
      int m0 = (idx >> 5) * 16;
      int nh0 = (idx & 31) * 16;           // column within H
      float t1[8];

      // pair 1: i (rows nh0) and g (rows 2*HDIM + nh0)
      {
        int ni = nh0, ngt = 2 * HDIM + nh0;
        v8f ai, ag;
        float bi = bias_g[ni + ln], bg = bias_g[ngt + ln];
#pragma unroll
        for (int r = 0; r < 8; ++r) { ai[r] = bi; ag[r] = bg; }
        for (int kb = 0; kb < FEAT; kb += 32) {
          v16bf a = load_a_lds(feat_bf, FEAT, m0, kb, lane);
          ai = wmma_bf16(a, load_b_w(Wih_bf, FEAT, ni, kb, lane), ai);
          ag = wmma_bf16(a, load_b_w(Wih_bf, FEAT, ngt, kb, lane), ag);
        }
        for (int kb = 0; kb < HDIM; kb += 32) {
          if (kb + 32 < HDIM)   // keep the W_hh stream hot (global_prefetch_b8)
            __builtin_prefetch(Whh_bf + (ni + ln) * HDIM + kb + 32, 0, 0);
          v16bf a = load_a_lds(h_bf, HDIM, m0, kb, lane);
          ai = wmma_bf16(a, load_b_w(Whh_bf, HDIM, ni, kb, lane), ai);
          ag = wmma_bf16(a, load_b_w(Whh_bf, HDIM, ngt, kb, lane), ag);
        }
#pragma unroll
        for (int r = 0; r < 8; ++r)
          t1[r] = fast_sigmoid(ai[r]) * fast_tanh(ag[r]);
      }

      // pair 2: f (rows HDIM + nh0) and o (rows 3*HDIM + nh0)
      {
        int nf = HDIM + nh0, no = 3 * HDIM + nh0;
        v8f af, ao;
        float bf = bias_g[nf + ln], bo = bias_g[no + ln];
#pragma unroll
        for (int r = 0; r < 8; ++r) { af[r] = bf; ao[r] = bo; }
        for (int kb = 0; kb < FEAT; kb += 32) {
          v16bf a = load_a_lds(feat_bf, FEAT, m0, kb, lane);
          af = wmma_bf16(a, load_b_w(Wih_bf, FEAT, nf, kb, lane), af);
          ao = wmma_bf16(a, load_b_w(Wih_bf, FEAT, no, kb, lane), ao);
        }
        for (int kb = 0; kb < HDIM; kb += 32) {
          v16bf a = load_a_lds(h_bf, HDIM, m0, kb, lane);
          af = wmma_bf16(a, load_b_w(Whh_bf, HDIM, nf, kb, lane), af);
          ao = wmma_bf16(a, load_b_w(Whh_bf, HDIM, no, kb, lane), ao);
        }
#pragma unroll
        for (int r = 0; r < 8; ++r) {
          float c2 = fast_sigmoid(af[r]) * c_acc[p][r] + t1[r];
          c_acc[p][r] = c2;
          h2v[p][r] = fast_sigmoid(ao[r]) * fast_tanh(c2);
        }
      }
    }
    __syncthreads();   // all reads of h_bf/dec_bf stable before updates

    // ---- Phase 3: commit h2; y = dec @ Wout^T + bout (pre-update h) ----
#pragma unroll
    for (int p = 0; p < 4; ++p) {
      int idx = wid * 4 + p;
      int m0 = (idx >> 5) * 16;
      int n = (idx & 31) * 16 + ln;
#pragma unroll
      for (int r = 0; r < 8; ++r)
        h_bf[(m0 + r + 8 * gg) * HDIM + n] = f32_to_bf16(h2v[p][r]);
    }
    if (wid < 16) {                        // 16 tiles: 4 x 4 over [64 x 64]
      int m0 = (wid >> 2) * 16, n0 = (wid & 3) * 16;
      int n = n0 + ln;
      v8f acc;
      float bn = bias_out[n];
#pragma unroll
      for (int r = 0; r < 8; ++r) acc[r] = bn;
      for (int kb = 0; kb < DEC; kb += 32)
        acc = wmma_bf16(load_a_lds(dec_bf, DEC, m0, kb, lane),
                        load_b_w(Wout_bf, DEC, n0, kb, lane), acc);
#pragma unroll
      for (int r = 0; r < 8; ++r) {
        int m = m0 + r + 8 * gg;
        float yv = acc[r];
        out[(t * BATCH + m) * XD + n] = yv;
        y_bf[m * XD + n] = f32_to_bf16(yv);
      }
    }
    __syncthreads();   // h_bf / y_bf visible for step t+1
  }
}

// ---------------- host-side launch ----------------
extern "C" void kernel_launch(void* const* d_in, const int* in_sizes, int n_in,
                              void* d_out, int out_size, void* d_ws, size_t ws_size,
                              hipStream_t stream) {
  const float* x    = (const float*)d_in[0];
  const float* mode = (const float*)d_in[1];
  const float* Wfx  = (const float*)d_in[2];
  const float* bfx  = (const float*)d_in[3];
  const float* Wih  = (const float*)d_in[4];
  const float* Whh  = (const float*)d_in[5];
  const float* bih  = (const float*)d_in[6];
  const float* bhh  = (const float*)d_in[7];
  const float* Whx  = (const float*)d_in[8];
  const float* bhx  = (const float*)d_in[9];
  const float* Wout = (const float*)d_in[10];
  const float* bout = (const float*)d_in[11];
  float* out = (float*)d_out;

  char* ws = (char*)d_ws;
  // workspace layout (bytes): all offsets 16B aligned
  unsigned short* Wfx_bf  = (unsigned short*)(ws + 0);        // 256*64*2    = 32768
  unsigned short* Wih_bf  = (unsigned short*)(ws + 32768);    // 2048*256*2  = 1048576
  unsigned short* Whh_bf  = (unsigned short*)(ws + 1081344);  // 2048*512*2  = 2097152
  unsigned short* Whx_bf  = (unsigned short*)(ws + 3178496);  // 256*512*2   = 262144
  unsigned short* Wout_bf = (unsigned short*)(ws + 3440640);  // 64*256*2    = 32768
  unsigned short* ftf_bf  = (unsigned short*)(ws + 3473408);  // 256*64*256*2= 8388608

  // 1) weight conversion (parallel)
  cvt_bf16_kernel<<<(FEAT * XD + 255) / 256, 256, 0, stream>>>(Wfx, Wfx_bf, FEAT * XD);
  cvt_bf16_kernel<<<(G4 * FEAT + 255) / 256, 256, 0, stream>>>(Wih, Wih_bf, G4 * FEAT);
  cvt_bf16_kernel<<<(G4 * HDIM + 255) / 256, 256, 0, stream>>>(Whh, Whh_bf, G4 * HDIM);
  cvt_bf16_kernel<<<(DEC * HDIM + 255) / 256, 256, 0, stream>>>(Whx, Whx_bf, DEC * HDIM);
  cvt_bf16_kernel<<<(XD * DEC + 255) / 256, 256, 0, stream>>>(Wout, Wout_bf, XD * DEC);

  // 2) teacher-forced features (parallel WMMA GEMM):
  //    16384x256 output tiles -> 16384 waves -> 2048 blocks of 8 waves
  feat_kernel<<<2048, 256, 0, stream>>>(x, Wfx_bf, bfx, ftf_bf);

  // 3) persistent sequential scan on one WGP
  scan_kernel<<<1, 1024, 0, stream>>>(Wfx_bf, Wih_bf, Whh_bf, Whx_bf, Wout_bf,
                                      ftf_bf, bfx, bih, bhh, bhx, bout, mode, out);
}
